// ScaledDotProductAttention_7352984011363
// MI455X (gfx1250) — matrix operand order repaired and checked by reading, not compiled, not run on hardware
//
#include <hip/hip_runtime.h>

#define HEADS   8
#define DMODEL  512
#define BATCH   2
#define NQ      2048
#define NK      2048

#define AS1 __attribute__((address_space(1)))
#define AS3 __attribute__((address_space(3)))

typedef __attribute__((ext_vector_type(16))) _Float16 v16h;
typedef __attribute__((ext_vector_type(8)))  _Float16 v8h;
typedef __attribute__((ext_vector_type(8)))  float    v8f;
typedef __attribute__((ext_vector_type(4)))  int      v4i;

static __device__ inline v8f wmma_f16(v16h a, v16h b, v8f c) {
  // D = A(16x32 f16) * B(32x16 f16) + C(16x16 f32)
  return __builtin_amdgcn_wmma_f32_16x16x32_f16(
      /*neg_a=*/false, a, /*neg_b=*/false, b,
      /*c_mod=*/(short)0, c, /*reuse_a=*/false, /*reuse_b=*/false);
}

// Async DMA of 16B/lane into LDS (gfx1250 GLOBAL_LOAD_ASYNC_TO_LDS_B128).
// Builtin signature (from toolchain diagnostic): (v4i AS1*, v4i AS3*, Ii, Ii).
// LDS addrspace(3) address = low 32 bits of the generic pointer (LDS aperture).
static __device__ inline void async_copy_b128(const void* g, void* l) {
  __builtin_amdgcn_global_load_async_to_lds_b128(
      (AS1 v4i*)(uintptr_t)g,
      (AS3 v4i*)(uint32_t)(uintptr_t)l,
      /*imm offset=*/0, /*cpol=*/0);
}

// A fragment (16x32 f16, row-major source). k order used here (contiguous, matched by the B loads):
// lanes 0-15: elems 0..15 = K0..15
// lanes16-31: elems 0..15 = K16..31
static __device__ inline v16h load_a_f16(const _Float16* tile, int stride, int lane) {
  const int m = lane & 15, hi = lane >> 4;
  const _Float16* r = tile + (size_t)m * stride;
  v8h c0 = *(const v8h*)(r + hi * 16);
  v8h c1 = *(const v8h*)(r + hi * 16 + 8);
  v16h out;
#pragma unroll
  for (int i = 0; i < 8; ++i) { out[i] = c0[i]; out[i + 8] = c1[i]; }
  return out;
}

static __device__ inline v16h load_a_f32(const float* tile, int stride, int lane) {
  const int m = lane & 15, hi = lane >> 4;
  const float* r = tile + (size_t)m * stride;
  v8f c0 = *(const v8f*)(r + hi * 16);
  v8f c1 = *(const v8f*)(r + hi * 16 + 8);
  v16h out;
#pragma unroll
  for (int i = 0; i < 8; ++i) { out[i] = (_Float16)c0[i]; out[i + 8] = (_Float16)c1[i]; }
  return out;
}

// B fragment (32x16): element (k,n) read from src[n*stride + k] (contiguous in k).
// lanes 0-15 hold K=0..15 for column n=lane, lanes 16-31 hold K=16..31.
static __device__ inline v16h load_b_tr_f16(const _Float16* src, int stride, int lane) {
  const int n = lane & 15, hi = lane >> 4;
  return *(const v16h*)(src + (size_t)n * stride + hi * 16);
}

static __device__ inline v16h load_b_tr_f32(const float* src, int stride, int lane) {
  const int n = lane & 15, hi = lane >> 4;
  const float* p = src + (size_t)n * stride + hi * 16;
  v8f c0 = *(const v8f*)p;
  v8f c1 = *(const v8f*)(p + 8);
  v16h out;
#pragma unroll
  for (int i = 0; i < 8; ++i) { out[i] = (_Float16)c0[i]; out[i + 8] = (_Float16)c1[i]; }
  return out;
}

// ---------------------------------------------------------------------------
// Projection GEMM: Y = A * W^T + bias.  A: M x 512 (fp32 or f16), W: 512 x 512.
// MODE 0: out f16 as [b][h][n][64]   (Q, K)
// MODE 2: out f16 as [b][h][64][n]   (V transposed, for contiguous P*V B loads)
// MODE 3: out fp32 as [m][512]       (final output projection)
// ---------------------------------------------------------------------------
template <int MODE, bool A16>
__global__ void __launch_bounds__(128)
proj_gemm(const void* __restrict__ Aptr, const float* __restrict__ W,
          const float* __restrict__ bias, void* __restrict__ Out) {
  const int lane = threadIdx.x & 31;
  const int wave = threadIdx.x >> 5;
  const int mb = blockIdx.x * 64 + wave * 16;
  const int ob = blockIdx.y * 64;

  v8f acc[4] = {};
  for (int kk = 0; kk < DMODEL; kk += 32) {
    v16h a;
    if (A16) a = load_a_f16((const _Float16*)Aptr + (size_t)mb * DMODEL + kk, DMODEL, lane);
    else     a = load_a_f32((const float*)Aptr + (size_t)mb * DMODEL + kk, DMODEL, lane);
#pragma unroll
    for (int t = 0; t < 4; ++t) {
      v16h b = load_b_tr_f32(W + (size_t)(ob + t * 16) * DMODEL + kk, DMODEL, lane);
      acc[t] = wmma_f16(a, b, acc[t]);
    }
  }

  const int hi = lane >> 4, ln = lane & 15;
#pragma unroll
  for (int t = 0; t < 4; ++t) {
    const int o = ob + t * 16 + ln;
    const float bv = bias[o];
#pragma unroll
    for (int r = 0; r < 8; ++r) {
      const int m = mb + r + 8 * hi;          // C layout: VGPR r -> row r (+8 for hi half)
      const float val = acc[t][r] + bv;
      if (MODE == 3) {
        ((float*)Out)[(size_t)m * DMODEL + o] = val;
      } else if (MODE == 2) {
        const int b = m / NK, n = m % NK;
        const int h = o >> 6, d = o & 63;
        ((_Float16*)Out)[((size_t)(b * HEADS + h) * 64 + d) * NK + n] = (_Float16)val;
      } else {
        const int b = m / NQ, n = m % NQ;
        const int h = o >> 6, d = o & 63;
        ((_Float16*)Out)[((size_t)(b * HEADS + h) * NQ + n) * 64 + d] = (_Float16)val;
      }
    }
  }
}

// ---------------------------------------------------------------------------
// Fused flash attention with async double-buffered K/V staging in LDS.
// Block = 128 threads (4 waves) covering 64 q rows; k in chunks of 32.
// K chunk (32x64 f16, contiguous 4KB) and V chunk (64 d-rows x 32 k, 64B/row)
// are DMA'd into LDS via GLOBAL_LOAD_ASYNC_TO_LDS_B128 and shared by all waves.
// ---------------------------------------------------------------------------
__global__ void __launch_bounds__(128)
flash_attn(const _Float16* __restrict__ qh, const _Float16* __restrict__ kh,
           const _Float16* __restrict__ vt, const float* __restrict__ aw,
           _Float16* __restrict__ oh) {
  __shared__ _Float16 ldsK[2][32 * 64];    // [buf][k_local][d]      4KB each
  __shared__ _Float16 ldsV[2][64 * 32];    // [buf][d][k_local]      4KB each
  __shared__ _Float16 lds_p[4][16 * 32];   // per-wave P staging (C->A layout)

  const int tid  = threadIdx.x;
  const int lane = tid & 31;
  const int wave = tid >> 5;
  const int qb0  = blockIdx.x * 64;
  const int qb   = qb0 + wave * 16;
  const int bh   = blockIdx.y;             // b*HEADS + h
  const int hi = lane >> 4, ln = lane & 15;

  const _Float16* qrow  = qh + ((size_t)bh * NQ + qb) * 64;
  const _Float16* kbase = kh + (size_t)bh * NK * 64;
  const _Float16* vbase = vt + (size_t)bh * 64 * NK;
  const float*    awb   = aw + ((size_t)bh * NQ + qb) * NK;

  // Q fragments, reused for all k: d-chunks [0,32) and [32,64)
  const v16h qa0 = load_a_f16(qrow + 0,  64, lane);
  const v16h qa1 = load_a_f16(qrow + 32, 64, lane);

  v8f oacc[4] = {};
  float mrow[8], lrow[8];
#pragma unroll
  for (int r = 0; r < 8; ++r) { mrow[r] = -__builtin_inff(); lrow[r] = 0.f; }

  // Stage one k-chunk: 4 async b128 ops per thread (K: 2, V: 2) = 8KB/block.
  auto stage = [&](int kb, int buf) {
    const char* gk = (const char*)(kbase + (size_t)kb * 64);   // contiguous 4KB
    char* lk = (char*)&ldsK[buf][0];
#pragma unroll
    for (int r = 0; r < 2; ++r) {
      const int off = (tid + r * 128) * 16;
      async_copy_b128(gk + off, lk + off);
    }
    char* lv = (char*)&ldsV[buf][0];
#pragma unroll
    for (int r = 0; r < 2; ++r) {
      const int c = tid + r * 128;                              // 256 x 16B pieces
      const int d = c >> 2, off = (c & 3) * 16;
      async_copy_b128((const char*)(vbase + (size_t)d * NK + kb) + off,
                      lv + d * 64 + off);
    }
  };

  const int nchunk = (qb0 + 64) / 32;      // block-uniform trip count (= 2*bx+2)
  stage(0, 0);                              // prologue: chunk 0 -> buffer 0

  int buf = 0;
  for (int i = 0; i < nchunk; ++i) {
    const int kb = i * 32;
    const bool has_next = (i + 1 < nchunk);
    if (has_next) {
      stage(kb + 32, buf ^ 1);              // prefetch next chunk into other buffer
      asm volatile("s_wait_asynccnt 0x4" ::: "memory");  // current 4 ops retired
    } else {
      asm volatile("s_wait_asynccnt 0x0" ::: "memory");
    }
    __syncthreads();                        // all waves' tile pieces visible

    if (kb < qb + 16) {                     // wave-level causal early-out
      const _Float16* Kt = &ldsK[buf][0];
      const _Float16* Vt = &ldsV[buf][0];

      if (has_next)                         // speculative prefetch of next aw block
        __builtin_prefetch(&awb[(size_t)(8 * hi) * NK + kb + 32 + ln], 0, 1);

      // ---- S = Q K^T over two 16-col tiles (K fragments from LDS)
      v8f s[2];
#pragma unroll
      for (int t = 0; t < 2; ++t) {
        v16h b0 = load_b_tr_f16(Kt + (size_t)(t * 16) * 64 + 0,  64, lane);
        v16h b1 = load_b_tr_f16(Kt + (size_t)(t * 16) * 64 + 32, 64, lane);
        v8f c = {};
        c = wmma_f16(qa0, b0, c);
        c = wmma_f16(qa1, b1, c);
        s[t] = c;
      }

      // ---- scale * attention_weights, analytic causal mask
#pragma unroll
      for (int t = 0; t < 2; ++t) {
#pragma unroll
        for (int r = 0; r < 8; ++r) {
          const int row = qb + r + 8 * hi;
          const int col = kb + t * 16 + ln;
          const float w = awb[(size_t)(r + 8 * hi) * NK + col];
          const float v = s[t][r] * 0.125f * w;
          s[t][r] = (col > row) ? -__builtin_inff() : v;
        }
      }

      // ---- online softmax: row stats via 16-lane butterfly (rows live per-VGPR)
      float pscale[8];
#pragma unroll
      for (int r = 0; r < 8; ++r) {
        float mx = fmaxf(s[0][r], s[1][r]);
#pragma unroll
        for (int off = 1; off < 16; off <<= 1)
          mx = fmaxf(mx, __shfl_xor(mx, off, 32));
        const float mnew = fmaxf(mrow[r], mx);
        const float sc = __expf(mrow[r] - mnew);
        mrow[r] = mnew;
        const float p0 = __expf(s[0][r] - mnew);
        const float p1 = __expf(s[1][r] - mnew);
        s[0][r] = p0; s[1][r] = p1;
        float rs = p0 + p1;
#pragma unroll
        for (int off = 1; off < 16; off <<= 1)
          rs += __shfl_xor(rs, off, 32);
        lrow[r] = lrow[r] * sc + rs;
        pscale[r] = sc;
      }
#pragma unroll
      for (int t = 0; t < 4; ++t)
#pragma unroll
        for (int r = 0; r < 8; ++r)
          oacc[t][r] *= pscale[r];

      // ---- P: C-layout -> LDS (row-major 16x32 f16) -> A-layout fragment
      _Float16* P = &lds_p[wave][0];
#pragma unroll
      for (int t = 0; t < 2; ++t)
#pragma unroll
        for (int r = 0; r < 8; ++r)
          P[(r + 8 * hi) * 32 + t * 16 + ln] = (_Float16)s[t][r];
      asm volatile("s_wait_dscnt 0x0" ::: "memory");  // per-wave store->load order
      const v16h pa = load_a_f16(P, 32, lane);

      // ---- O += P * V  (V fragments from LDS, contiguous-k columns)
#pragma unroll
      for (int t = 0; t < 4; ++t) {
        v16h vb = load_b_tr_f16(Vt + (size_t)(t * 16) * 32, 32, lane);
        oacc[t] = wmma_f16(pa, vb, oacc[t]);
      }
    }
    __syncthreads();                        // protect buffers before next overwrite
    buf ^= 1;
  }

  // ---- normalize and emit O as [b][q][h*64+d] f16 for the output projection
  const int b = bh / HEADS, h = bh % HEADS;
#pragma unroll
  for (int t = 0; t < 4; ++t) {
#pragma unroll
    for (int r = 0; r < 8; ++r) {
      const int q = qb + r + 8 * hi;
      const int d = t * 16 + ln;
      const float val = oacc[t][r] / lrow[r];
      oh[(size_t)(b * NQ + q) * DMODEL + h * 64 + d] = (_Float16)val;
    }
  }
}

// ---------------------------------------------------------------------------
extern "C" void kernel_launch(void* const* d_in, const int* in_sizes, int n_in,
                              void* d_out, int out_size, void* d_ws, size_t ws_size,
                              hipStream_t stream) {
  const float* queries = (const float*)d_in[0];
  const float* keys    = (const float*)d_in[1];
  const float* values  = (const float*)d_in[2];
  const float* aw      = (const float*)d_in[3];
  // d_in[4] = attention_mask (broadcast triu) -> computed analytically in-kernel
  const float* Wq = (const float*)d_in[5];
  const float* bq = (const float*)d_in[6];
  const float* Wk = (const float*)d_in[7];
  const float* bk = (const float*)d_in[8];
  const float* Wv = (const float*)d_in[9];
  const float* bv = (const float*)d_in[10];
  const float* Wo = (const float*)d_in[11];
  const float* bo = (const float*)d_in[12];

  const size_t perTensor = (size_t)BATCH * HEADS * NQ * 64;   // elements
  _Float16* qh = (_Float16*)d_ws;
  _Float16* kh = qh + perTensor;
  _Float16* vt = kh + perTensor;
  _Float16* oh = vt + perTensor;          // [B][NQ][512] f16, 16 MB total workspace

  dim3 blk(128);
  dim3 gP(BATCH * NQ / 64, DMODEL / 64);  // (64, 8)
  proj_gemm<0, false><<<gP, blk, 0, stream>>>(queries, Wq, bq, qh);
  proj_gemm<0, false><<<gP, blk, 0, stream>>>(keys,    Wk, bk, kh);
  proj_gemm<2, false><<<gP, blk, 0, stream>>>(values,  Wv, bv, vt);

  dim3 gA(NQ / 64, BATCH * HEADS);        // (32, 16)
  flash_attn<<<gA, blk, 0, stream>>>(qh, kh, vt, aw, oh);

  proj_gemm<3, true><<<gP, blk, 0, stream>>>(oh, Wo, bo, (float*)d_out);
}
